// MAB_PRE_GPTJ_90443421319311
// MI455X (gfx1250) — matrix operand order repaired and checked by reading, not compiled, not run on hardware
//
#include <hip/hip_runtime.h>
#include <hip/hip_bf16.h>

#define BB 2
#define SQ 2048
#define SK 2048
#define DD 1024
#define HH 16
#define DH 64
#define EPS 1e-5f

typedef __attribute__((ext_vector_type(16))) __bf16 v16bf;
typedef __attribute__((ext_vector_type(8)))  __bf16 bfx8;
typedef __attribute__((ext_vector_type(8)))  float  v8f;

// ---------------------------------------------------------------------------
// WMMA helpers (CDNA5 wave32 bf16, f32 accumulate)
// ---------------------------------------------------------------------------
__device__ __forceinline__ v8f wmma_bf16(v16bf a, v16bf b, v8f c) {
  return __builtin_amdgcn_wmma_f32_16x16x32_bf16(false, a, false, b, (short)0, c,
                                                 false, false);
}

// A fragment (16x32 bf16): lane = M row; K interleave per ISA layout:
//   lanes 0-15: K = 0..7 and 16..23 ; lanes 16-31: K = 8..15 and 24..31
__device__ __forceinline__ v16bf load_a_frag(const __bf16* rowptr, int k0, int half) {
  union { v16bf v; bfx8 h[2]; } u;
  const __bf16* p = rowptr + k0 + half * 8;
  u.h[0] = *(const bfx8*)(p);
  u.h[1] = *(const bfx8*)(p + 16);
  return u.v;
}

// B fragment (32x16 bf16): lane = N column; K interleave per ISA layout:
//   lanes 0-15: K = 0..7 and 16..23 ; lanes 16-31: K = 8..15 and 24..31;
// source is contraction-major with row stride ld -> strided gather.
__device__ __forceinline__ v16bf load_b_frag(const __bf16* Y, int ld, int l15,
                                             int half, int k0, int n0) {
  union { v16bf v; __bf16 e[16]; } u;
#pragma unroll
  for (int s = 0; s < 16; ++s)
    u.e[s] = Y[(size_t)(k0 + 8 * half + (s & 7) + 16 * (s >> 3)) * ld + n0 + l15];
  return u.v;
}

// ---------------------------------------------------------------------------
// Kernel 0: weight convert + transpose: Wt[d][j] = (bf16)W[j][d]
// ---------------------------------------------------------------------------
__global__ __launch_bounds__(256) void wt_kernel(const float* __restrict__ W,
                                                 __bf16* __restrict__ Wt) {
  __shared__ float tile[32][33];
  const int bx = blockIdx.x * 32;  // j
  const int by = blockIdx.y * 32;  // d
  const int tx = threadIdx.x, ty = threadIdx.y;  // 32 x 8
#pragma unroll
  for (int i = 0; i < 32; i += 8)
    tile[ty + i][tx] = W[(size_t)(bx + ty + i) * DD + by + tx];
  __syncthreads();
#pragma unroll
  for (int i = 0; i < 32; i += 8)
    Wt[(size_t)(by + ty + i) * DD + bx + tx] = (__bf16)tile[tx][ty + i];
}

// ---------------------------------------------------------------------------
// Kernel 1: row LayerNorm (f32 in) -> bf16 out.  One block per row of 1024.
// ---------------------------------------------------------------------------
__global__ __launch_bounds__(256) void ln_bf16_kernel(const float* __restrict__ X,
                                                      const float* __restrict__ g,
                                                      const float* __restrict__ bt,
                                                      __bf16* __restrict__ Y) {
  __shared__ float red[256];
  const int row = blockIdx.x, t = threadIdx.x;
  const float* x = X + (size_t)row * DD;
  float v[4];
  float s = 0.f;
#pragma unroll
  for (int i = 0; i < 4; ++i) { v[i] = x[i * 256 + t]; s += v[i]; }
  red[t] = s; __syncthreads();
  for (int o = 128; o > 0; o >>= 1) { if (t < o) red[t] += red[t + o]; __syncthreads(); }
  const float mean = red[0] * (1.f / DD);
  __syncthreads();
  float ss = 0.f;
#pragma unroll
  for (int i = 0; i < 4; ++i) { const float d = v[i] - mean; ss += d * d; }
  red[t] = ss; __syncthreads();
  for (int o = 128; o > 0; o >>= 1) { if (t < o) red[t] += red[t + o]; __syncthreads(); }
  const float rstd = rsqrtf(red[0] * (1.f / DD) + EPS);
  __bf16* y = Y + (size_t)row * DD;
#pragma unroll
  for (int i = 0; i < 4; ++i) {
    const int c = i * 256 + t;
    y[c] = (__bf16)((v[i] - mean) * rstd * g[c] + bt[c]);
  }
}

// ---------------------------------------------------------------------------
// Kernel 2: C = X(bf16, 4096x1024) . W^T (+bias).  Wt stored d-major bf16.
// Block = 256 thr (8 waves); wave computes a 32(M) x 64(N) tile (2 A-frags
// sharing 4 B-frags -> 8 WMMA per 32-k chunk, 24 B of loads per WMMA).
// MODE: 0=Q1 [b][h][sq][dh] bf16 *pre-scaled by 1/sqrt(D)*,
//       1=K1t [b][h][dh][sk] bf16, 2=V1 [b][h][sk][dh] bf16,
//       3=O2 relu f32 [row][col]
// ---------------------------------------------------------------------------
template <int MODE>
__global__ __launch_bounds__(256) void gemm_kernel(const __bf16* __restrict__ X,
                                                   const __bf16* __restrict__ Wt,
                                                   const float* __restrict__ bias,
                                                   void* __restrict__ out) {
  const int lane = threadIdx.x & 31;
  const int wave = threadIdx.x >> 5;
  const int l15 = lane & 15;
  const int half = lane >> 4;
  const int m0 = blockIdx.x * 256 + wave * 32;
  const int n0 = blockIdx.y * 64;
  const __bf16* Xrow0 = X + (size_t)(m0 + l15) * DD;
  const __bf16* Xrow1 = Xrow0 + (size_t)16 * DD;
  v8f c[2][4] = {{v8f{}, v8f{}, v8f{}, v8f{}}, {v8f{}, v8f{}, v8f{}, v8f{}}};
  for (int kk = 0; kk < DD; kk += 32) {
    const v16bf a0 = load_a_frag(Xrow0, kk, half);
    const v16bf a1 = load_a_frag(Xrow1, kk, half);
#pragma unroll
    for (int j = 0; j < 4; ++j) {
      const v16bf bf = load_b_frag(Wt, DD, l15, half, kk, n0 + j * 16);
      c[0][j] = wmma_bf16(a0, bf, c[0][j]);
      c[1][j] = wmma_bf16(a1, bf, c[1][j]);
    }
  }
  const int h = n0 >> 6;
#pragma unroll
  for (int mt = 0; mt < 2; ++mt) {
#pragma unroll
    for (int j = 0; j < 4; ++j) {
      const int n = j * 16 + l15;  // column within the 64-wide head tile
      const float bsum = bias[n0 + n];
#pragma unroll
      for (int r = 0; r < 8; ++r) {
        const int m = m0 + mt * 16 + r + 8 * half;  // global row (b*2048 + s)
        const int bb = m >> 11;
        const int sr = m & 2047;
        const float val = c[mt][j][r] + bsum;
        if (MODE == 0) {        // Q1, folded 1/sqrt(D) = 1/32 score scale
          ((__bf16*)out)[(((size_t)(bb * HH + h)) * SQ + sr) * DH + n] =
              (__bf16)(val * 0.03125f);
        } else if (MODE == 1) { // K1 transposed per head
          ((__bf16*)out)[(((size_t)(bb * HH + h)) * DH + n) * SK + sr] = (__bf16)val;
        } else if (MODE == 2) { // V1
          ((__bf16*)out)[(((size_t)(bb * HH + h)) * SK + sr) * DH + n] = (__bf16)val;
        } else {                // O2 = relu(...)
          ((float*)out)[(size_t)m * DD + n0 + n] = fmaxf(val, 0.f);
        }
      }
    }
  }
}

// ---------------------------------------------------------------------------
// Kernel 3: flash attention per (b,h).  Block = 128 thr (4 waves); each wave
// owns 16 q-rows x DH=64.  SK swept in 32-wide chunks with online softmax.
// Row-sums are folded into a 5th WMMA accumulator S = sum(P . ones).
// Scores arrive pre-scaled (1/32 folded into Q1).
// ---------------------------------------------------------------------------
__global__ __launch_bounds__(128) void attn_kernel(const __bf16* __restrict__ Q1,
                                                   const __bf16* __restrict__ K1t,
                                                   const __bf16* __restrict__ V1,
                                                   const int* __restrict__ pad,
                                                   float* __restrict__ O1) {
  __shared__ __align__(16) __bf16 Plds[4][16 * 32];
  const int lane = threadIdx.x & 31;
  const int wave = threadIdx.x >> 5;
  const int l15 = lane & 15;
  const int half = lane >> 4;
  const int bh = blockIdx.x;
  const int b = bh >> 4;
  const int h = bh & 15;
  const int q0 = blockIdx.y * 64 + wave * 16;

  const __bf16* Qb = Q1 + ((size_t)bh * SQ + q0 + l15) * DH;
  const v16bf a0 = load_a_frag(Qb, 0, half);
  const v16bf a1 = load_a_frag(Qb, 32, half);
  const __bf16* Kh = K1t + (size_t)bh * DH * SK;
  const __bf16* Vh = V1 + (size_t)bh * SK * DH;
  const int* pm = pad + b * SK;

  // ones B-fragment for the row-sum accumulator
  union { v16bf v; __bf16 e[16]; } uo;
#pragma unroll
  for (int i = 0; i < 16; ++i) uo.e[i] = (__bf16)1.0f;
  const v16bf ones = uo.v;

  v8f acc[4] = {v8f{}, v8f{}, v8f{}, v8f{}};
  v8f ssum = v8f{};
  float mrow[8];
#pragma unroll
  for (int r = 0; r < 8; ++r) mrow[r] = -1e30f;
  __bf16* Pw = &Plds[wave][0];

  for (int kc = 0; kc < SK; kc += 32) {
    // score tiles: S(16q x 32k) = Q(16x64) . K^T ; contraction DH=64 -> 2 wmma each
    v8f s0 = v8f{}, s1 = v8f{};
    s0 = wmma_bf16(a0, load_b_frag(Kh, SK, l15, half, 0, kc), s0);
    s0 = wmma_bf16(a1, load_b_frag(Kh, SK, l15, half, 32, kc), s0);
    s1 = wmma_bf16(a0, load_b_frag(Kh, SK, l15, half, 0, kc + 16), s1);
    s1 = wmma_bf16(a1, load_b_frag(Kh, SK, l15, half, 32, kc + 16), s1);

    if (kc + 64 <= SK - 32) {  // prefetch chunk two ahead (global_prefetch_b8 path)
      __builtin_prefetch(Kh + (size_t)(l15 + 16 * half) * SK + kc + 64, 0, 1);
      __builtin_prefetch(Vh + (size_t)(kc + 64 + l15 + 16 * half) * DH, 0, 1);
    }

    const bool ok0 = pm[kc + l15] != 0;
    const bool ok1 = pm[kc + 16 + l15] != 0;

    asm volatile("" ::: "memory");  // keep this iter's LDS writes below prior reads
#pragma unroll
    for (int r = 0; r < 8; ++r) {
      float x0 = ok0 ? s0[r] : -1e12f;
      float x1 = ok1 ? s1[r] : -1e12f;
      float mx = fmaxf(x0, x1);
      mx = fmaxf(mx, __shfl_xor(mx, 1, 32));
      mx = fmaxf(mx, __shfl_xor(mx, 2, 32));
      mx = fmaxf(mx, __shfl_xor(mx, 4, 32));
      mx = fmaxf(mx, __shfl_xor(mx, 8, 32));
      const float mnew = fmaxf(mrow[r], mx);
      const float f = __expf(mrow[r] - mnew);
      const float p0 = ok0 ? __expf(x0 - mnew) : 0.f;
      const float p1 = ok1 ? __expf(x1 - mnew) : 0.f;
      mrow[r] = mnew;
      ssum[r] *= f;
#pragma unroll
      for (int j = 0; j < 4; ++j) acc[j][r] *= f;
      const int m = r + 8 * half;                 // D-layout row
      Pw[m * 32 + l15] = (__bf16)p0;
      Pw[m * 32 + 16 + l15] = (__bf16)p1;
    }
    asm volatile("s_wait_dscnt 0" ::: "memory");  // wave-private LDS: writes done

    const v16bf pf = load_a_frag(Pw + l15 * 32, 0, half);
    ssum = wmma_bf16(pf, ones, ssum);             // row-sum via matrix pipe
#pragma unroll
    for (int j = 0; j < 4; ++j) {
      const v16bf vf = load_b_frag(Vh, DH, l15, half, kc, j * 16);
      acc[j] = wmma_bf16(pf, vf, acc[j]);
    }
  }

#pragma unroll
  for (int r = 0; r < 8; ++r) {
    const float l = ssum[r];                      // every column holds the row sum
    const float inv = l > 0.f ? __builtin_amdgcn_rcpf(l) : 0.f;
    const int q = q0 + r + 8 * half;
    float* o = O1 + ((size_t)(b * SQ + q)) * DD + h * DH;
#pragma unroll
    for (int j = 0; j < 4; ++j) o[j * 16 + l15] = acc[j][r] * inv;
  }
}

// ---------------------------------------------------------------------------
// Kernel 4: out = LN(Q + O1 + O2) with ln_f
// ---------------------------------------------------------------------------
__global__ __launch_bounds__(256) void final_ln_kernel(const float* __restrict__ Q,
                                                       const float* __restrict__ O1,
                                                       const float* __restrict__ O2,
                                                       const float* __restrict__ g,
                                                       const float* __restrict__ bt,
                                                       float* __restrict__ out) {
  __shared__ float red[256];
  const int row = blockIdx.x, t = threadIdx.x;
  const size_t base = (size_t)row * DD;
  float v[4];
  float s = 0.f;
#pragma unroll
  for (int i = 0; i < 4; ++i) {
    const int c = i * 256 + t;
    v[i] = Q[base + c] + O1[base + c] + O2[base + c];
    s += v[i];
  }
  red[t] = s; __syncthreads();
  for (int o = 128; o > 0; o >>= 1) { if (t < o) red[t] += red[t + o]; __syncthreads(); }
  const float mean = red[0] * (1.f / DD);
  __syncthreads();
  float ss = 0.f;
#pragma unroll
  for (int i = 0; i < 4; ++i) { const float d = v[i] - mean; ss += d * d; }
  red[t] = ss; __syncthreads();
  for (int o = 128; o > 0; o >>= 1) { if (t < o) red[t] += red[t + o]; __syncthreads(); }
  const float rstd = rsqrtf(red[0] * (1.f / DD) + EPS);
#pragma unroll
  for (int i = 0; i < 4; ++i) {
    const int c = i * 256 + t;
    out[base + c] = (v[i] - mean) * rstd * g[c] + bt[c];
  }
}

// ---------------------------------------------------------------------------
extern "C" void kernel_launch(void* const* d_in, const int* in_sizes, int n_in,
                              void* d_out, int out_size, void* d_ws, size_t ws_size,
                              hipStream_t stream) {
  (void)in_sizes; (void)n_in; (void)out_size; (void)ws_size;
  const float* Q   = (const float*)d_in[0];
  const float* K   = (const float*)d_in[1];
  const int*  pad  = (const int*)d_in[2];
  const float* Wq  = (const float*)d_in[3];
  const float* Wk  = (const float*)d_in[4];
  const float* Wv  = (const float*)d_in[5];
  const float* Wo  = (const float*)d_in[6];
  const float* bq  = (const float*)d_in[7];
  const float* bk  = (const float*)d_in[8];
  const float* bv  = (const float*)d_in[9];
  const float* bo  = (const float*)d_in[10];
  const float* lqg = (const float*)d_in[11];
  const float* lqb = (const float*)d_in[12];
  const float* lkg = (const float*)d_in[13];
  const float* lkb = (const float*)d_in[14];
  const float* lfg = (const float*)d_in[15];
  const float* lfb = (const float*)d_in[16];
  float* out = (float*)d_out;

  char* ws = (char*)d_ws;
  size_t off = 0;
  auto alloc = [&](size_t bytes) {
    char* p = ws + off;
    off += (bytes + 255) & ~(size_t)255;
    return p;
  };
  const size_t ROWS = (size_t)BB * SQ;                 // 4096
  __bf16* Qn  = (__bf16*)alloc(ROWS * DD * 2);
  __bf16* Kn  = (__bf16*)alloc(ROWS * DD * 2);
  __bf16* Wqt = (__bf16*)alloc((size_t)DD * DD * 2);
  __bf16* Wkt = (__bf16*)alloc((size_t)DD * DD * 2);
  __bf16* Wvt = (__bf16*)alloc((size_t)DD * DD * 2);
  __bf16* Wot = (__bf16*)alloc((size_t)DD * DD * 2);
  __bf16* Q1  = (__bf16*)alloc((size_t)BB * HH * SQ * DH * 2);
  __bf16* K1t = (__bf16*)alloc((size_t)BB * HH * DH * SK * 2);
  __bf16* V1  = (__bf16*)alloc((size_t)BB * HH * SK * DH * 2);
  float*  O2  = (float*)alloc(ROWS * DD * 4);
  float*  O1b = (float*)alloc(ROWS * DD * 4);

  const dim3 tgrid(DD / 32, DD / 32), tblk(32, 8);
  wt_kernel<<<tgrid, tblk, 0, stream>>>(Wq, Wqt);
  wt_kernel<<<tgrid, tblk, 0, stream>>>(Wk, Wkt);
  wt_kernel<<<tgrid, tblk, 0, stream>>>(Wv, Wvt);
  wt_kernel<<<tgrid, tblk, 0, stream>>>(Wo, Wot);

  ln_bf16_kernel<<<ROWS, 256, 0, stream>>>(Q, lqg, lqb, Qn);
  ln_bf16_kernel<<<ROWS, 256, 0, stream>>>(K, lkg, lkb, Kn);

  const dim3 ggrid(ROWS / 256, DD / 64);
  gemm_kernel<0><<<ggrid, 256, 0, stream>>>(Qn, Wqt, bq, (void*)Q1);
  gemm_kernel<1><<<ggrid, 256, 0, stream>>>(Kn, Wkt, bk, (void*)K1t);
  gemm_kernel<2><<<ggrid, 256, 0, stream>>>(Kn, Wvt, bv, (void*)V1);
  gemm_kernel<3><<<ggrid, 256, 0, stream>>>(Qn, Wot, bo, (void*)O2);

  attn_kernel<<<dim3(BB * HH, SQ / 64), 128, 0, stream>>>(Q1, K1t, V1, pad, O1b);

  final_ln_kernel<<<ROWS, 256, 0, stream>>>(Q, O1b, O2, lfg, lfb, out);
}
